// OverlapTripletLoss_11991548690925
// MI455X (gfx1250) — compile-verified, matrix operand order checked by reading, not run on hardware
//
#include <hip/hip_runtime.h>
#include <hip/hip_bf16.h>

typedef __attribute__((ext_vector_type(16))) __bf16        v16bf;
typedef __attribute__((ext_vector_type(8)))  float         v8f;
typedef __attribute__((ext_vector_type(8)))  unsigned int  v8u;
typedef __attribute__((ext_vector_type(4)))  unsigned int  v4u;

#define N_TOT 200000
#define C_CLS 100
#define D_DIM 256
#define M_PER 2000
#define K_SEL 100
#define C_PAD 112      // centers padded to 7x16 tiles
#define LDSTR 264      // padded LDS row stride (bf16 elems) -> conflict-free

__device__ __forceinline__ unsigned short f2bf_hi(float f) {
    unsigned u = __float_as_uint(f);
    unsigned r = u + 0x7FFFu + ((u >> 16) & 1u);   // RNE to bf16
    return (unsigned short)(r >> 16);
}
__device__ __forceinline__ float bfu2f(unsigned short h) {
    return __uint_as_float(((unsigned)h) << 16);
}

// ---------------- Kernel 1: per-class centers (f32 mean) -> bf16 hi/lo + ||c||^2
__global__ void centers_kernel(const float* __restrict__ x,
                               unsigned short* __restrict__ chi,
                               unsigned short* __restrict__ clo,
                               float* __restrict__ c2n) {
    int c = blockIdx.x;            // 0..111 (rows >=100 are zero padding)
    int d = threadIdx.x;           // 0..255
    float val = 0.0f;
    if (c < C_CLS) {
        const float* row = x + (size_t)c * M_PER * D_DIM + d;
        float s = 0.0f;
        for (int m = 0; m < M_PER; ++m) s += row[(size_t)m * D_DIM];
        val = s * (1.0f / (float)M_PER);
    }
    unsigned short h = f2bf_hi(val);
    float lo = val - bfu2f(h);
    chi[c * D_DIM + d] = h;
    clo[c * D_DIM + d] = f2bf_hi(lo);

    __shared__ float red[256];
    red[d] = val * val;
    __syncthreads();
    for (int s = 128; s > 0; s >>= 1) {
        if (d < s) red[d] += red[d + s];
        __syncthreads();
    }
    if (d == 0) c2n[c] = red[0];
}

// ---------------- Kernel 2: WMMA bf16x2 GEMM (centers x x^T) fused with distance
__global__ void __launch_bounds__(256)
gemm_dist_kernel(const float* __restrict__ x,
                 const unsigned short* __restrict__ chi,
                 const unsigned short* __restrict__ clo,
                 const float* __restrict__ c2n,
                 float* __restrict__ dist) {
    extern __shared__ char smem[];
    unsigned short* s_chi = (unsigned short*)smem;           // [112][264]
    unsigned short* s_clo = s_chi + C_PAD * LDSTR;           // [112][264]
    unsigned short* s_xhi = s_clo + C_PAD * LDSTR;           // [32][264]
    unsigned short* s_xlo = s_xhi + 32 * LDSTR;              // [32][264]
    float*          s_x2  = (float*)(s_xlo + 32 * LDSTR);    // [32]
    float*          s_c2  = s_x2 + 32;                       // [112]

    const int tid    = threadIdx.x;
    const int n_base = blockIdx.x * 32;                      // N_TOT % 32 == 0

    // stage centers (hi/lo) into LDS with padded stride
    for (int idx = tid; idx < C_PAD * D_DIM; idx += 256) {
        int r = idx >> 8, c = idx & 255;
        s_chi[r * LDSTR + c] = chi[idx];
        s_clo[r * LDSTR + c] = clo[idx];
    }
    if (tid < C_PAD) s_c2[tid] = c2n[tid];
    if (tid < 32)    s_x2[tid] = 0.0f;
    __syncthreads();

    // stage 32x256 x-tile -> bf16 hi/lo, accumulate row ||x||^2
    {
        int r = tid >> 3;                 // row in tile 0..31 (8 threads/row)
        int q = tid & 7;
        const float4* xr = (const float4*)(x + (size_t)(n_base + r) * D_DIM);
        float ss = 0.0f;
        #pragma unroll
        for (int k = 0; k < 8; ++k) {
            int c4 = q + k * 8;           // float4 index 0..63, coalesced
            float4 v = xr[c4];
            float f[4] = {v.x, v.y, v.z, v.w};
            #pragma unroll
            for (int j = 0; j < 4; ++j) {
                float fv = f[j];
                ss += fv * fv;
                unsigned short h = f2bf_hi(fv);
                s_xhi[r * LDSTR + c4 * 4 + j] = h;
                s_xlo[r * LDSTR + c4 * 4 + j] = f2bf_hi(fv - bfu2f(h));
            }
        }
        atomicAdd(&s_x2[r], ss);
    }
    __syncthreads();

    const int wave = tid >> 5;
    const int lane = tid & 31;      // wave32
    const int half = lane >> 4;
    const int mn   = lane & 15;

    // 7 center-tiles x 2 sample-tiles = 14 wave tasks over 8 waves
    for (int task = wave; task < 14; task += 8) {
        int c_tile = task % 7;
        int n_tile = task / 7;
        int c_row  = c_tile * 16 + mn;     // A-matrix row (per-lane)
        int x_row  = n_tile * 16 + mn;     // B-matrix column (per-lane)
        v8f acc = {};
        #pragma unroll
        for (int kb = 0; kb < 8; ++kb) {   // K = 256 in 8 steps of 32
            // per-lane fragment = two contiguous 8-bf16 (16 B) runs:
            //   k = koff + 0..7   and   k = koff + 16..23
            int koff = kb * 32 + half * 8;             // ushort offset, 16B-aligned
            const v4u* pah = (const v4u*)(s_chi + c_row * LDSTR + koff);
            const v4u* pal = (const v4u*)(s_clo + c_row * LDSTR + koff);
            const v4u* pbh = (const v4u*)(s_xhi + x_row * LDSTR + koff);
            const v4u* pbl = (const v4u*)(s_xlo + x_row * LDSTR + koff);
            v8u ah = __builtin_shufflevector(pah[0], pah[2], 0, 1, 2, 3, 4, 5, 6, 7);
            v8u al = __builtin_shufflevector(pal[0], pal[2], 0, 1, 2, 3, 4, 5, 6, 7);
            v8u bh = __builtin_shufflevector(pbh[0], pbh[2], 0, 1, 2, 3, 4, 5, 6, 7);
            v8u bl = __builtin_shufflevector(pbl[0], pbl[2], 0, 1, 2, 3, 4, 5, 6, 7);
            v16bf a_hi = __builtin_bit_cast(v16bf, ah);
            v16bf a_lo = __builtin_bit_cast(v16bf, al);
            v16bf b_hi = __builtin_bit_cast(v16bf, bh);
            v16bf b_lo = __builtin_bit_cast(v16bf, bl);
            // bf16x2 split: hi*hi + hi*lo + lo*hi  (~f32 accuracy)
            acc = __builtin_amdgcn_wmma_f32_16x16x32_bf16(false, a_hi, false, b_hi,
                                                          (short)0, acc, false, false);
            acc = __builtin_amdgcn_wmma_f32_16x16x32_bf16(false, a_hi, false, b_lo,
                                                          (short)0, acc, false, false);
            acc = __builtin_amdgcn_wmma_f32_16x16x32_bf16(false, a_lo, false, b_hi,
                                                          (short)0, acc, false, false);
        }
        // cx -> distance, store. D layout: lane n = lane%16, vgpr r -> m = r + 8*half
        float x2v = s_x2[n_tile * 16 + mn];
        #pragma unroll
        for (int r = 0; r < 8; ++r) {
            int m  = r + 8 * half;
            int c1 = c_tile * 16 + m;
            if (c1 < C_CLS) {
                float d2 = s_c2[c1] + x2v - 2.0f * acc[r];
                dist[(size_t)c1 * N_TOT + n_base + n_tile * 16 + mn] =
                    sqrtf(fmaxf(d2, 0.0f));
            }
        }
    }
}

// ---------------- Kernel 3: per-(c1,c2) top-K selection via LDS bitonic sort
__global__ void select_kernel(const float* __restrict__ dist,
                              float* __restrict__ pos_mean,
                              float* __restrict__ neg_mean) {
    __shared__ float v[2048];
    __shared__ float red[128];
    const int c1 = blockIdx.x, c2 = blockIdx.y;
    const int tid = threadIdx.x;
    const bool diag = (c1 == c2);
    const float pad = diag ? -__builtin_inff() : __builtin_inff();
    const float* src = dist + (size_t)c1 * N_TOT + (size_t)c2 * M_PER;

    for (int i = tid; i < 2048; i += 256) v[i] = (i < M_PER) ? src[i] : pad;
    __syncthreads();

    for (unsigned k = 2; k <= 2048; k <<= 1) {
        for (unsigned j = k >> 1; j > 0; j >>= 1) {
            for (unsigned i = tid; i < 2048; i += 256) {
                unsigned ixj = i ^ j;
                if (ixj > i) {
                    float a = v[i], b = v[ixj];
                    bool up = ((i & k) == 0);
                    if (up ? (a > b) : (a < b)) { v[i] = b; v[ixj] = a; }
                }
            }
            __syncthreads();
        }
    }
    // diagonal: mean of 100 largest; off-diagonal: mean of 100 smallest
    int base = diag ? (2048 - K_SEL) : 0;
    if (tid < 128) red[tid] = (tid < K_SEL) ? v[base + tid] : 0.0f;
    __syncthreads();
    for (int s = 64; s > 0; s >>= 1) {
        if (tid < s) red[tid] += red[tid + s];
        __syncthreads();
    }
    if (tid == 0) {
        float mean = red[0] * (1.0f / (float)K_SEL);
        if (diag) { pos_mean[c1] = mean; neg_mean[c1 * C_CLS + c2] = 0.0f; }
        else        neg_mean[c1 * C_CLS + c2] = mean;
    }
}

// ---------------- Kernel 4: masked hinge sum -> scalar loss
__global__ void loss_kernel(const float* __restrict__ pos_mean,
                            const float* __restrict__ neg_mean,
                            float* __restrict__ out) {
    __shared__ float red[256];
    int tid = threadIdx.x;
    float s = 0.0f;
    for (int idx = tid; idx < C_CLS * C_CLS; idx += 256) {
        int c1 = idx / C_CLS, c2 = idx % C_CLS;
        if (c1 != c2)
            s += fmaxf(1.0f + pos_mean[c1] - neg_mean[idx], 0.0f);
    }
    red[tid] = s;
    __syncthreads();
    for (int st = 128; st > 0; st >>= 1) {
        if (tid < st) red[tid] += red[tid + st];
        __syncthreads();
    }
    if (tid == 0) out[0] = red[0] / ((float)(C_CLS + 1) * ((float)C_CLS * 0.5f));
}

extern "C" void kernel_launch(void* const* d_in, const int* in_sizes, int n_in,
                              void* d_out, int out_size, void* d_ws, size_t ws_size,
                              hipStream_t stream) {
    const float* x = (const float*)d_in[0];   // [N, D] f32; d_in[1] (y) implicit by layout

    char* ws = (char*)d_ws;
    const size_t SZ_DIST = (size_t)C_CLS * N_TOT * sizeof(float);     // 80 MB
    const size_t OFF_CHI = SZ_DIST;
    const size_t OFF_CLO = OFF_CHI + (size_t)C_PAD * D_DIM * 2;
    const size_t OFF_C2  = OFF_CLO + (size_t)C_PAD * D_DIM * 2;
    const size_t OFF_POS = OFF_C2  + 512;
    const size_t OFF_NEG = OFF_POS + 512;
    const size_t NEED    = OFF_NEG + (size_t)C_CLS * C_CLS * sizeof(float);
    if (ws_size < NEED) return;

    float*          dist = (float*)(ws);
    unsigned short* chi  = (unsigned short*)(ws + OFF_CHI);
    unsigned short* clo  = (unsigned short*)(ws + OFF_CLO);
    float*          c2n  = (float*)(ws + OFF_C2);
    float*          posm = (float*)(ws + OFF_POS);
    float*          negm = (float*)(ws + OFF_NEG);

    centers_kernel<<<C_PAD, 256, 0, stream>>>(x, chi, clo, c2n);

    size_t shmem = (size_t)(2 * C_PAD * LDSTR + 2 * 32 * LDSTR) * sizeof(unsigned short)
                 + (size_t)(32 + C_PAD) * sizeof(float);              // ~152.6 KB < 320 KB/WGP
    gemm_dist_kernel<<<N_TOT / 32, 256, shmem, stream>>>(x, chi, clo, c2n, dist);

    dim3 grid_sel(C_CLS, C_CLS);
    select_kernel<<<grid_sel, 256, 0, stream>>>(dist, posm, negm);

    loss_kernel<<<1, 256, 0, stream>>>(posm, negm, (float*)d_out);
}
